// M_Baseline_LSTM_80238579024390
// MI455X (gfx1250) — compile-verified, matrix operand order checked by reading, not run on hardware
//
#include <hip/hip_runtime.h>
#include <math.h>

typedef __bf16 bf16;
typedef __attribute__((ext_vector_type(16))) __bf16 v16bf;
typedef __attribute__((ext_vector_type(8)))  __bf16 v8bf;
typedef __attribute__((ext_vector_type(8)))  float  v8f;

#define B_    64
#define S_    256
#define H_    512
#define G4H   2048
#define D_PAD 320
#define HSTRIDE 520   // LDS row stride (bf16 elems), padded to dodge bank conflicts

__device__ __forceinline__ float sigmoidf_(float x) { return 1.0f / (1.0f + __expf(-x)); }

__device__ __forceinline__ v16bf join16(v8bf lo, v8bf hi) {
  return __builtin_shufflevector(lo, hi, 0,1,2,3,4,5,6,7,8,9,10,11,12,13,14,15);
}

// Async copy 16B global -> LDS (tracked by ASYNCcnt), gfx1250.
__device__ __forceinline__ void async_load_b128(unsigned lds_off, const void* gptr) {
  asm volatile("global_load_async_to_lds_b128 %0, %1, off"
               :: "v"(lds_off), "v"((unsigned long long)(uintptr_t)gptr)
               : "memory");
}
__device__ __forceinline__ void wait_async0() {
  asm volatile("s_wait_asynccnt 0" ::: "memory");
}

// ---------------------------------------------------------------------------
// Pack a (Ksrc x N) row-major f32 weight into bf16 WMMA-B fragment blocks.
// Block (kt,nt) covers K=[kt*32,+32) x N=[nt*16,+16); 512 bf16 per block,
// lane l owns 16 contiguous bf16 at block+l*16. Fragment convention:
// lane<16: N=l,  K = {0..7, 16..23} ; lane>=16: N=l-16, K = {8..15, 24..31}.
// Rows k >= Ksrc are zero (pads W0's K=300 up to 320).
// ---------------------------------------------------------------------------
__global__ void pack_b(const float* __restrict__ src, bf16* __restrict__ dst,
                       int Ksrc, int K, int N)
{
  int o = blockIdx.x * blockDim.x + threadIdx.x;
  if (o >= K * N) return;
  int j    = o & 15;
  int l    = (o >> 4) & 31;
  int blk  = o >> 9;
  int ntiles = N >> 4;
  int nt = blk % ntiles;
  int kt = blk / ntiles;
  int base8 = (l < 16) ? 0 : 8;
  int k = kt * 32 + base8 + ((j < 8) ? j : (j + 8));
  int n = nt * 16 + (l & 15);
  float v = (k < Ksrc) ? src[(size_t)k * N + n] : 0.0f;
  dst[o] = (bf16)v;
}

// ---------------------------------------------------------------------------
// Embedding gather + zero-pad D=300 -> 320, f32 -> bf16.
// ---------------------------------------------------------------------------
__global__ void embed_pad(const int* __restrict__ text, const float* __restrict__ emb,
                          bf16* __restrict__ xpad)
{
  long long o = (long long)blockIdx.x * blockDim.x + threadIdx.x;
  long long total = (long long)B_ * S_ * D_PAD;
  if (o >= total) return;
  int col = (int)(o % D_PAD);
  long long row = o / D_PAD;          // token index b*S + s
  float v = 0.0f;
  if (col < 300) {
    int tok = text[row];
    v = emb[(long long)tok * 300 + col];
  }
  xpad[o] = (bf16)v;
}

// ---------------------------------------------------------------------------
// Tiled bf16 WMMA GEMM: C[M x 2048](bf16) = A[M x K](bf16,row-major) @ Bpack + bias
// Block = 256 threads (8 waves). Block tile = 128 rows x 256 cols.
// B panel staged through LDS with double-buffered async global->LDS copies
// (fetched once per block instead of once per wave => 8x less L2 traffic);
// compute on buffer k overlaps the async fill of buffer k+1. B fragments are
// software-prefetched one tile ahead so each v_wmma waits dscnt<=1, not 0.
// ---------------------------------------------------------------------------
__global__ void __launch_bounds__(256)
gemm_bf16(const bf16* __restrict__ A, int lda, int ktiles,
          const bf16* __restrict__ Bp, int ntiles_total,
          const float* __restrict__ bias,
          bf16* __restrict__ C, int ldc)
{
  __shared__ bf16 Bs[2][16 * 512];          // 2 x 16KB

  const int tid   = threadIdx.x;
  const int lane  = tid & 31;
  const int w     = tid >> 5;
  const int n16   = lane & 15;
  const int hi    = lane >> 4;
  const int base8 = hi << 3;
  const int rowbase = blockIdx.y * 128 + w * 16;
  const int colblk  = blockIdx.x;           // 256-col block
  const int ntblk   = colblk * 16;

  auto issue = [&](int kr, int buf) {
    const bf16* src = Bp + (((size_t)kr * ntiles_total + ntblk) << 9);
    unsigned dst = (unsigned)(uintptr_t)(&Bs[buf][0]);
#pragma unroll
    for (int i = 0; i < 4; ++i) {
      int off = (tid + i * 256) * 8;        // bf16 elements, 16B per thread-chunk
      async_load_b128(dst + (unsigned)off * 2u, src + off);
    }
  };

  v8f acc[16] = {};
  const bf16* aptr = A + (size_t)(rowbase + n16) * lda + base8;

  issue(0, 0);
  for (int kr = 0; kr < ktiles; ++kr) {
    const int cb = kr & 1;
    wait_async0();                          // my async fills of Bs[cb] done
    __syncthreads();                        // everyone's done; Bs[cb^1] free
    if (kr + 1 < ktiles) issue(kr + 1, cb ^ 1);

    v8bf alo = *(const v8bf*)(aptr + kr * 32);
    v8bf ahi = *(const v8bf*)(aptr + kr * 32 + 16);
    v16bf a = join16(alo, ahi);

    const bf16* bsp = &Bs[cb][0] + lane * 16;
    v16bf bcur = *(const v16bf*)(bsp);      // prefetch tile 0
#pragma unroll
    for (int t = 0; t < 16; ++t) {
      v16bf bnext = bcur;
      if (t < 15) bnext = *(const v16bf*)(bsp + (t + 1) * 512);   // prefetch t+1
      acc[t] = __builtin_amdgcn_wmma_f32_16x16x32_bf16(false, a, false, bcur,
                                                       (short)0, acc[t], false, false);
      bcur = bnext;
    }
  }

#pragma unroll
  for (int t = 0; t < 16; ++t) {
    int col = colblk * 256 + t * 16 + n16;
    float bv = bias[col];
#pragma unroll
    for (int vr = 0; vr < 8; ++vr) {
      int row = rowbase + vr + 8 * hi;      // C/D layout: lanes 16-31 hold M+8
      C[(size_t)row * ldc + col] = (bf16)(acc[t][vr] + bv);
    }
  }
}

// ---------------------------------------------------------------------------
// Persistent LSTM recurrence. grid = 2 blocks (0=fwd, 1=bwd), 1024 threads.
// h kept double-buffered in LDS (bf16, 2 x 64 x 520 = 130 KB of the 320 KB
// WGP LDS); h & c also carried in VGPRs (each wave owns the same 4 tiles of
// h every step -> no cross-wave hazard on c). One __syncthreads per step.
// U fragments are wave-unique, so direct global (L2-resident) loads are
// already minimal traffic. 256 WMMA ops per wave per step.
// ---------------------------------------------------------------------------
__global__ void __launch_bounds__(1024)
lstm_rec(const bf16* __restrict__ xzf, const bf16* __restrict__ xzb,
         const bf16* __restrict__ Uf,  const bf16* __restrict__ Ub,
         const int* __restrict__ text,
         bf16* __restrict__ out_seq,   // layer0: (B,S,1024) bf16; layer1: null
         float* __restrict__ out_hT)   // layer1: (B,1024) f32;   layer0: null
{
  __shared__ bf16 hbuf[2][B_ * HSTRIDE];

  const int dir = blockIdx.x;                 // 0 = forward, 1 = backward
  const bf16* xz = dir ? xzb : xzf;
  const bf16* Up = dir ? Ub : Uf;
  const int coloff = dir * H_;

  const int tid   = threadIdx.x;
  const int lane  = tid & 31;
  const int w     = tid >> 5;                 // wave id 0..31 == h col-tile
  const int n16   = lane & 15;
  const int hi    = lane >> 4;
  const int base8 = hi << 3;

  for (int i = tid; i < B_ * HSTRIDE; i += 1024) hbuf[0][i] = (bf16)0.0f;
  __syncthreads();

  v8f c_reg[4] = {};
  v8f h_reg[4] = {};

  int cur = 0;
  for (int t = 0; t < S_; ++t) {
    const int s = dir ? (S_ - 1 - t) : t;     // original sequence index

#pragma unroll
    for (int rt = 0; rt < 4; ++rt) {
      v8f acc[4];
      // init accumulators with precomputed xz (= x@W + b) gate tiles
#pragma unroll
      for (int g = 0; g < 4; ++g) {
#pragma unroll
        for (int vr = 0; vr < 8; ++vr) {
          int b = rt * 16 + vr + 8 * hi;
          int col = g * H_ + w * 16 + n16;
          acc[g][vr] = (float)xz[((size_t)b * S_ + s) * G4H + col];
        }
      }
      // gates += h @ U  (K = 512, 16 WMMA K-steps, A shared across 4 gates)
      const bf16* hb = hbuf[cur] + (rt * 16 + n16) * HSTRIDE + base8;
#pragma unroll 4
      for (int kr = 0; kr < 16; ++kr) {
        v8bf alo = *(const v8bf*)(hb + kr * 32);
        v8bf ahi = *(const v8bf*)(hb + kr * 32 + 16);
        v16bf a = join16(alo, ahi);
#pragma unroll
        for (int g = 0; g < 4; ++g) {
          int gct = g * 32 + w;
          v16bf b = *(const v16bf*)(Up + (((size_t)kr * 128 + gct) << 9) + lane * 16);
          acc[g] = __builtin_amdgcn_wmma_f32_16x16x32_bf16(false, a, false, b,
                                                           (short)0, acc[g], false, false);
        }
      }
      // elementwise LSTM cell + mask carry
#pragma unroll
      for (int vr = 0; vr < 8; ++vr) {
        int b = rt * 16 + vr + 8 * hi;
        bool m = (text[b * S_ + s] != 0);
        float iv = sigmoidf_(acc[0][vr]);
        float fv = sigmoidf_(acc[1][vr]);
        float gv = tanhf(acc[2][vr]);
        float ov = sigmoidf_(acc[3][vr]);
        float cn = fv * c_reg[rt][vr] + iv * gv;
        float hn = ov * tanhf(cn);
        if (m) { c_reg[rt][vr] = cn; h_reg[rt][vr] = hn; }
        int hcol = w * 16 + n16;
        hbuf[cur ^ 1][b * HSTRIDE + hcol] = (bf16)h_reg[rt][vr];
        if (out_seq)
          out_seq[((size_t)b * S_ + s) * 1024 + coloff + hcol] = (bf16)h_reg[rt][vr];
      }
    }
    __syncthreads();
    cur ^= 1;
  }

  if (out_hT) {
#pragma unroll
    for (int rt = 0; rt < 4; ++rt)
#pragma unroll
      for (int vr = 0; vr < 8; ++vr) {
        int b = rt * 16 + vr + 8 * hi;
        int hcol = w * 16 + n16;
        out_hT[(size_t)b * 1024 + coloff + hcol] = h_reg[rt][vr];
      }
  }
}

// ---------------------------------------------------------------------------
// Small f32 dense layer (M<=64): one thread per output element.
// ---------------------------------------------------------------------------
__global__ void dense_f32(const float* __restrict__ X, const float* __restrict__ W,
                          const float* __restrict__ bias, float* __restrict__ Y,
                          int M, int K, int N, int leaky)
{
  int o = blockIdx.x * blockDim.x + threadIdx.x;
  if (o >= M * N) return;
  int n = o % N, m = o / N;
  float s = bias[n];
  for (int k = 0; k < K; ++k) s += X[(size_t)m * K + k] * W[(size_t)k * N + n];
  if (leaky && s < 0.0f) s *= 0.2f;
  Y[o] = s;
}

// ---------------------------------------------------------------------------
extern "C" void kernel_launch(void* const* d_in, const int* in_sizes, int n_in,
                              void* d_out, int out_size, void* d_ws, size_t ws_size,
                              hipStream_t stream)
{
  (void)in_sizes; (void)n_in; (void)out_size; (void)ws_size;

  const int*   text = (const int*)  d_in[0];
  const float* emb  = (const float*)d_in[1];
  const float* W0f = (const float*)d_in[2];  const float* U0f = (const float*)d_in[3];  const float* b0f = (const float*)d_in[4];
  const float* W0b = (const float*)d_in[5];  const float* U0b = (const float*)d_in[6];  const float* b0b = (const float*)d_in[7];
  const float* W1f = (const float*)d_in[8];  const float* U1f = (const float*)d_in[9];  const float* b1f = (const float*)d_in[10];
  const float* W1b = (const float*)d_in[11]; const float* U1b = (const float*)d_in[12]; const float* b1b = (const float*)d_in[13];
  const float* d0w = (const float*)d_in[14]; const float* d0b = (const float*)d_in[15];
  const float* d1w = (const float*)d_in[16]; const float* d1b = (const float*)d_in[17];
  const float* d2w = (const float*)d_in[18]; const float* d2b = (const float*)d_in[19];
  float* out = (float*)d_out;

  char* wsp = (char*)d_ws;
  size_t off = 0;
  auto alloc = [&](size_t bytes) -> char* {
    char* p = wsp + off;
    off += (bytes + 255) & ~(size_t)255;
    return p;
  };
  bf16* xpad = (bf16*)alloc((size_t)B_ * S_ * D_PAD * 2);
  bf16* xzf  = (bf16*)alloc((size_t)B_ * S_ * G4H * 2);   // reused for layer-1 fwd xz
  bf16* xzb  = (bf16*)alloc((size_t)B_ * S_ * G4H * 2);   // reused for layer-1 bwd xz
  bf16* l1in = (bf16*)alloc((size_t)B_ * S_ * 1024 * 2);
  bf16* pw0f = (bf16*)alloc((size_t)D_PAD * G4H * 2);
  bf16* pw0b = (bf16*)alloc((size_t)D_PAD * G4H * 2);
  bf16* pu0f = (bf16*)alloc((size_t)H_ * G4H * 2);
  bf16* pu0b = (bf16*)alloc((size_t)H_ * G4H * 2);
  bf16* pu1f = (bf16*)alloc((size_t)H_ * G4H * 2);
  bf16* pu1b = (bf16*)alloc((size_t)H_ * G4H * 2);
  bf16* pw1f = (bf16*)alloc((size_t)1024 * G4H * 2);
  bf16* pw1b = (bf16*)alloc((size_t)1024 * G4H * 2);
  float* hT  = (float*)alloc((size_t)B_ * 1024 * 4);
  float* t0  = (float*)alloc((size_t)B_ * 512 * 4);
  float* t1  = (float*)alloc((size_t)B_ * 256 * 4);

  const int th = 256;
  auto gp = [&](long long total) { return dim3((unsigned)((total + th - 1) / th)); };

  // 1) pack weights into WMMA B-fragment blocks (bf16)
  pack_b<<<gp((long long)D_PAD * G4H), th, 0, stream>>>(W0f, pw0f, 300, D_PAD, G4H);
  pack_b<<<gp((long long)D_PAD * G4H), th, 0, stream>>>(W0b, pw0b, 300, D_PAD, G4H);
  pack_b<<<gp((long long)H_ * G4H),    th, 0, stream>>>(U0f, pu0f, H_, H_, G4H);
  pack_b<<<gp((long long)H_ * G4H),    th, 0, stream>>>(U0b, pu0b, H_, H_, G4H);
  pack_b<<<gp((long long)H_ * G4H),    th, 0, stream>>>(U1f, pu1f, H_, H_, G4H);
  pack_b<<<gp((long long)H_ * G4H),    th, 0, stream>>>(U1b, pu1b, H_, H_, G4H);
  pack_b<<<gp((long long)1024 * G4H),  th, 0, stream>>>(W1f, pw1f, 1024, 1024, G4H);
  pack_b<<<gp((long long)1024 * G4H),  th, 0, stream>>>(W1b, pw1b, 1024, 1024, G4H);

  // 2) embedding gather + pad to bf16
  embed_pad<<<gp((long long)B_ * S_ * D_PAD), th, 0, stream>>>(text, emb, xpad);

  // 3) layer-0 input projections (WMMA GEMM, M=16384, K=320, N=2048)
  dim3 ggrid(G4H / 256, (B_ * S_) / 128);
  gemm_bf16<<<ggrid, 256, 0, stream>>>(xpad, D_PAD, D_PAD / 32, pw0f, G4H / 16, b0f, xzf, G4H);
  gemm_bf16<<<ggrid, 256, 0, stream>>>(xpad, D_PAD, D_PAD / 32, pw0b, G4H / 16, b0b, xzb, G4H);

  // 4) layer-0 recurrences (fwd + bwd concurrently), write (B,S,1024) bf16 seq
  lstm_rec<<<2, 1024, 0, stream>>>(xzf, xzb, pu0f, pu0b, text, l1in, (float*)nullptr);

  // 5) layer-1 input projections (K=1024), reuse xz buffers
  gemm_bf16<<<ggrid, 256, 0, stream>>>(l1in, 1024, 1024 / 32, pw1f, G4H / 16, b1f, xzf, G4H);
  gemm_bf16<<<ggrid, 256, 0, stream>>>(l1in, 1024, 1024 / 32, pw1b, G4H / 16, b1b, xzb, G4H);

  // 6) layer-1 recurrences, only final hT (B,1024) f32
  lstm_rec<<<2, 1024, 0, stream>>>(xzf, xzb, pu1f, pu1b, text, (bf16*)nullptr, hT);

  // 7) MLP head (tiny, f32)
  dense_f32<<<gp((long long)B_ * 512), th, 0, stream>>>(hT, d0w, d0b, t0, B_, 1024, 512, 1);
  dense_f32<<<gp((long long)B_ * 256), th, 0, stream>>>(t0, d1w, d1b, t1, B_, 512, 256, 1);
  dense_f32<<<gp((long long)B_),       th, 0, stream>>>(t1, d2w, d2b, out, B_, 256, 1, 0);
}
